// AlignmentAttentionLayer_71219147702755
// MI455X (gfx1250) — matrix operand order repaired and checked by reading, not compiled, not run on hardware
//
#include <hip/hip_runtime.h>
#include <hip/hip_bf16.h>

// ---------------------------------------------------------------------------
// AlignmentAttentionLayer on MI455X (gfx1250, wave32, WMMA).
//
//   h = tanh(W_p * (Y @ softmax(w^T tanh(W_y Y + W_h h_n e^T))) + W_x h_n)
//
// All dense GEMMs use v_wmma_f32_16x16x32_f16 (f16 operands, f32 accumulate).
// Weights / h_n / r are pre-converted to f16 zero-padded to DP=320 so padded
// K columns contribute exact zeros.  The dominant kernel (attn) fuses:
//   - async staging of Y[b] (GLOBAL_LOAD_ASYNC_TO_LDS_B128, ASYNCcnt) double-
//     buffered against the WMMA loop (falls back to sync loads if the
//     builtins are unavailable),
//   - f32->f16 transpose to the B-fragment layout in LDS,
//   - W_y*Y GEMM with W_y A-fragments preloaded in VGPRs (invariant across
//     the batch loop -> W_y is read from L2 once per workgroup),
//   - bias + tanh + w-dot via LDS ds_add_f32,
//   - softmax with wave32 shfl_xor reductions,
//   - r = Y*alpha, stored f16 for the final WMMA GEMM.
//
// Workspace (~74 MB of d_ws) laid out below.
// ---------------------------------------------------------------------------

#define DD    300
#define LL    50
#define BATCH 16384
#define DP    320          // padded D (20 WMMA tiles)
#define LP    64           // padded L (4 WMMA tiles)
#define YSTRIDE 328        // LDS row stride in halfwords: 656B = 164 DW, 164%64=36
                           // -> 16 lanes land on 16 distinct 4-bank groups
#define BT    8            // batch rows per workgroup in attn kernel
#define ATTN_WAVES 10
#define ATTN_THREADS (ATTN_WAVES * 32)

#define YS_BYTES  (LP * YSTRIDE * 2)   // 41984
#define RAW_BYTES (DD * LL * 4)        // 60000
#define DYN_LDS   (YS_BYTES + RAW_BYTES)

typedef __attribute__((ext_vector_type(16))) _Float16 v16h;
typedef __attribute__((ext_vector_type(8)))  _Float16 v8h;
typedef __attribute__((ext_vector_type(8)))  float    v8f;

#if defined(__gfx1250__) && __has_builtin(__builtin_amdgcn_global_load_async_to_lds_b128) && __has_builtin(__builtin_amdgcn_s_wait_asynccnt)
#define USE_ASYNC 1
#else
#define USE_ASYNC 0
#endif

#if USE_ASYNC
// The builtin's params are pointers to a 16-byte int vector in addrspace(1)
// (printed by clang as "__device__") and addrspace(3) ("__shared__").
typedef int v4i_vs __attribute__((vector_size(4 * sizeof(int))));
typedef __attribute__((address_space(1))) v4i_vs gv4i;
typedef __attribute__((address_space(3))) v4i_vs lv4i;

// One 16-byte async copy global -> LDS per enabled lane (tracked by ASYNCcnt).
// AS casts go through integers: flat LDS addresses carry the LDS byte offset
// in their low 32 bits (ISA 10.2 aperture rules).
__device__ __forceinline__ void async_copy16(const void* g, void* l) {
    __builtin_amdgcn_global_load_async_to_lds_b128(
        (gv4i*)(unsigned long long)(size_t)g,
        (lv4i*)(unsigned int)(size_t)l, 0, 0);
}
#endif

// A fragment (16x32 f16, M x K): lane holds row (lane&15); 16 contiguous
// K values starting at koff = (lane>>4)*16.
__device__ __forceinline__ v16h load_a_frag(const _Float16* p) {
    return *(const v16h*)p;
}

// B fragment (32x16 f16, K x N): lane holds column (lane&15); 16 contiguous
// K values starting at koff = (lane>>4)*16.  Requires [N][K]-contiguous data.
__device__ __forceinline__ v16h load_b_frag(const _Float16* p) {
    return *(const v16h*)p;
}

__device__ __forceinline__ v8f wmma_f16(v16h a, v16h b, v8f c) {
    return __builtin_amdgcn_wmma_f32_16x16x32_f16(false, a, false, b, (short)0, c, false, false);
}

// ---------------------------------------------------------------------------
// Kernel 0: convert weights + h_n to zero-padded f16, pad w.
// ---------------------------------------------------------------------------
__global__ __launch_bounds__(256) void prep_kernel(
    const float* __restrict__ Wy, const float* __restrict__ Wh,
    const float* __restrict__ Wp, const float* __restrict__ Wx,
    const float* __restrict__ hn, const float* __restrict__ w,
    _Float16* __restrict__ wy16, _Float16* __restrict__ wh16,
    _Float16* __restrict__ wp16, _Float16* __restrict__ wx16,
    _Float16* __restrict__ hn16, float* __restrict__ wpad)
{
    const size_t i = (size_t)blockIdx.x * blockDim.x + threadIdx.x;

    if (i < (size_t)4 * DP * DP) {
        const int which = (int)(i / (DP * DP));
        const int rem   = (int)(i % (DP * DP));
        const int d = rem / DP, e = rem % DP;
        const float* src = (which == 0) ? Wy : (which == 1) ? Wh : (which == 2) ? Wp : Wx;
        _Float16*    dst = (which == 0) ? wy16 : (which == 1) ? wh16 : (which == 2) ? wp16 : wx16;
        const float v = (d < DD && e < DD) ? src[d * DD + e] : 0.0f;
        dst[rem] = (_Float16)v;
    }
    if (i < (size_t)BATCH * DP) {
        const int b = (int)(i / DP), e = (int)(i % DP);
        hn16[i] = (_Float16)((e < DD) ? hn[(size_t)b * DD + e] : 0.0f);
    }
    if (i < DP) {
        wpad[i] = (i < DD) ? w[i] : 0.0f;
    }
}

// ---------------------------------------------------------------------------
// Kernel A: Z[b][m] = sum_e W16[m][e] * hn16[b][e]   (M=320, N=16384, K=320)
// ---------------------------------------------------------------------------
__global__ __launch_bounds__(256) void gemm_hn_kernel(
    const _Float16* __restrict__ W16, const _Float16* __restrict__ hn16,
    float* __restrict__ Z)
{
    const int gwave = (int)((blockIdx.x * blockDim.x + threadIdx.x) >> 5);
    const int lane  = threadIdx.x & 31;
    const int mt = gwave / (BATCH / 16);   // 0..19
    const int nt = gwave % (BATCH / 16);   // 0..1023
    const int r_ = lane & 15;
    const int hi = lane >> 4;

    v8f c = {};
    const _Float16* arow  = W16 + (size_t)(mt * 16 + r_) * DP + hi * 16;
    const _Float16* bcolp = hn16 + (size_t)(nt * 16 + r_) * DP + hi * 16;
#pragma unroll
    for (int k = 0; k < DP; k += 32) {
        v16h a = load_a_frag(arow + k);
        v16h b = load_b_frag(bcolp + k);
        c = wmma_f16(a, b, c);
    }
    const int bcol = nt * 16 + r_;
    float* zp = Z + (size_t)bcol * DP + mt * 16 + hi * 8;
#pragma unroll
    for (int v = 0; v < 8; ++v) zp[v] = c[v];
}

// ---------------------------------------------------------------------------
// Kernel B (main attention).
// ---------------------------------------------------------------------------
__global__ __launch_bounds__(ATTN_THREADS) void attn_kernel(
    const float* __restrict__ Y, const float* __restrict__ Z1,
    const _Float16* __restrict__ wy16, const float* __restrict__ wpad,
    _Float16* __restrict__ r16)
{
    extern __shared__ char dynsmem[];
    _Float16* Ys  = (_Float16*)dynsmem;                 // [LP][YSTRIDE] f16
    float*    raw = (float*)(dynsmem + YS_BYTES);       // DD*LL f32 bounce buffer
    __shared__ float score[LP];
    __shared__ float alpha[LP];
    __shared__ float z1s[DP];
    __shared__ float wls[DP];

    const int tid  = threadIdx.x;
    const int lane = tid & 31;
    const int wave = tid >> 5;       // 0..9
    const int r_   = lane & 15;
    const int hi   = lane >> 4;

    // ---- preload this wave's W_y A-fragments: 2 M-tiles x 10 K-steps.
    // Invariant across the batch loop -> W_y leaves L2 once per workgroup.
    v16h afr[10][2];
    {
        const _Float16* a0 = wy16 + (size_t)(wave * 32 + r_) * DP + hi * 16;
        const _Float16* a1 = a0 + (size_t)16 * DP;
#pragma unroll
        for (int kk = 0; kk < 10; ++kk) {
            afr[kk][0] = load_a_frag(a0 + kk * 32);
            afr[kk][1] = load_a_frag(a1 + kk * 32);
        }
    }
    if (tid < DP) wls[tid] = wpad[tid];

    // Zero LDS pad regions once: K-pad columns (e>=300) must be exact zero;
    // rows l>=50 zeroed so discarded output columns never see NaNs.
    for (int i = tid; i < LP * YSTRIDE; i += ATTN_THREADS) {
        const int l = i / YSTRIDE, e = i % YSTRIDE;
        if (l >= LL || e >= DD) Ys[i] = (_Float16)0.0f;
    }

#if USE_ASYNC
    {   // kick off async staging of the first batch row
        const float* Yb0 = Y + (size_t)(blockIdx.x * BT) * (DD * LL);
        for (int ch = tid; ch < (DD * LL) / 4; ch += ATTN_THREADS)
            async_copy16(Yb0 + ch * 4, raw + ch * 4);
    }
#endif
    __syncthreads();

    for (int bb = 0; bb < BT; ++bb) {
        const int b = blockIdx.x * BT + bb;
        const float* Yb = Y + (size_t)b * (DD * LL);

        // ---- stage Y[b] (f32 [e][l]) -> LDS f16 [l][e] ----
#if USE_ASYNC
        __builtin_amdgcn_s_wait_asynccnt(0);
        __syncthreads();                      // all waves' async chunks landed
        for (int i = tid; i < DD * LL; i += ATTN_THREADS) {
            const int e = i / LL, l = i - e * LL;
            Ys[l * YSTRIDE + e] = (_Float16)raw[i];
        }
#else
        for (int i = tid; i < DD * LL; i += ATTN_THREADS) {
            const int e = i / LL, l = i - e * LL;
            Ys[l * YSTRIDE + e] = (_Float16)Yb[i];
        }
        if (bb + 1 < BT) {
            const float* Yn = Yb + DD * LL;
            const int pi = tid * 48;
            if (pi < DD * LL) __builtin_prefetch(Yn + pi, 0, 1);
        }
#endif
        if (tid < LP) score[tid] = 0.0f;
        if (tid < DP) z1s[tid] = Z1[(size_t)b * DP + tid];
        __syncthreads();                      // Ys/z1s ready; raw free again

#if USE_ASYNC
        if (bb + 1 < BT) {                    // overlap next row's HBM fetch
            const float* Yn = Yb + DD * LL;   // with this row's WMMA loop
            for (int ch = tid; ch < (DD * LL) / 4; ch += ATTN_THREADS)
                async_copy16(Yn + ch * 4, raw + ch * 4);
        }
#endif

        // ---- GEMM: wave owns M-tiles 2*wave, 2*wave+1 x all 4 N-tiles ----
        v8f c[2][4] = {};
#pragma unroll
        for (int kk = 0; kk < 10; ++kk) {
#pragma unroll
            for (int j = 0; j < 4; ++j) {
                v16h bf = load_b_frag(Ys + (j * 16 + r_) * YSTRIDE + kk * 32 + hi * 16);
                c[0][j] = wmma_f16(afr[kk][0], bf, c[0][j]);
                c[1][j] = wmma_f16(afr[kk][1], bf, c[1][j]);
            }
        }

        // ---- bias + tanh + w-dot (lane's column n = j*16 + r_) ----
        const int m0 = wave * 32;
        float part[4] = {0.0f, 0.0f, 0.0f, 0.0f};
#pragma unroll
        for (int i = 0; i < 2; ++i) {
#pragma unroll
            for (int v = 0; v < 8; ++v) {
                const int m = m0 + i * 16 + hi * 8 + v;
                const float bias = z1s[m];
                const float wm   = wls[m];           // zero for padded rows
#pragma unroll
                for (int j = 0; j < 4; ++j)
                    part[j] += wm * tanhf(c[i][j][v] + bias);
            }
        }
#pragma unroll
        for (int j = 0; j < 4; ++j)
            atomicAdd(&score[j * 16 + r_], part[j]);   // ds_add_f32
        __syncthreads();

        // ---- softmax over l in [0,50): wave 0, wave32 shfl reductions ----
        if (wave == 0) {
            float s0 = (lane < LL)      ? score[lane]      : -1e30f;
            float s1 = (lane + 32 < LL) ? score[lane + 32] : -1e30f;
            float mx = fmaxf(s0, s1);
#pragma unroll
            for (int off = 16; off > 0; off >>= 1)
                mx = fmaxf(mx, __shfl_xor(mx, off, 32));
            float e0 = (lane < LL)      ? __expf(s0 - mx) : 0.0f;
            float e1 = (lane + 32 < LL) ? __expf(s1 - mx) : 0.0f;
            float sm = e0 + e1;
#pragma unroll
            for (int off = 16; off > 0; off >>= 1)
                sm += __shfl_xor(sm, off, 32);
            const float inv = 1.0f / sm;
            alpha[lane]      = e0 * inv;   // zero for lane in [50,64)
            alpha[lane + 32] = e1 * inv;
        }
        __syncthreads();

        // ---- r[b][d] = sum_l alpha[l] * Ys[l][d]; store f16 [b][e] ----
        {
            const int d = tid;             // ATTN_THREADS == DP
            float acc = 0.0f;
            if (d < DD) {
#pragma unroll 10
                for (int l = 0; l < LL; ++l)
                    acc += alpha[l] * (float)Ys[l * YSTRIDE + d];
            }
            r16[(size_t)b * DP + d] = (_Float16)acc;
        }
        __syncthreads();   // protect Ys before next iteration overwrites it
    }
}

// ---------------------------------------------------------------------------
// Kernel C: out[b][m] = tanh( sum_e W_p[m][e] r[b][e] + Z2[b][m] ), m < 300
// ---------------------------------------------------------------------------
__global__ __launch_bounds__(256) void gemm_out_kernel(
    const _Float16* __restrict__ wp16, const _Float16* __restrict__ r16,
    const float* __restrict__ Z2, float* __restrict__ out)
{
    const int gwave = (int)((blockIdx.x * blockDim.x + threadIdx.x) >> 5);
    const int lane  = threadIdx.x & 31;
    const int mt = gwave / (BATCH / 16);
    const int nt = gwave % (BATCH / 16);
    const int r_ = lane & 15;
    const int hi = lane >> 4;

    v8f c = {};
    const _Float16* arow  = wp16 + (size_t)(mt * 16 + r_) * DP + hi * 16;
    const _Float16* bcolp = r16 + (size_t)(nt * 16 + r_) * DP + hi * 16;
#pragma unroll
    for (int k = 0; k < DP; k += 32) {
        v16h a = load_a_frag(arow + k);
        v16h b = load_b_frag(bcolp + k);
        c = wmma_f16(a, b, c);
    }
    const int bcol = nt * 16 + r_;
    const float* z2 = Z2 + (size_t)bcol * DP;
#pragma unroll
    for (int v = 0; v < 8; ++v) {
        const int m = mt * 16 + hi * 8 + v;
        if (m < DD)
            out[(size_t)bcol * DD + m] = tanhf(c[v] + z2[m]);
    }
}

// ---------------------------------------------------------------------------
// Host-side launch
// ---------------------------------------------------------------------------
extern "C" void kernel_launch(void* const* d_in, const int* in_sizes, int n_in,
                              void* d_out, int out_size, void* d_ws, size_t ws_size,
                              hipStream_t stream) {
    (void)in_sizes; (void)n_in; (void)out_size; (void)ws_size;

    const float* Y  = (const float*)d_in[0];
    const float* hn = (const float*)d_in[1];
    const float* Wy = (const float*)d_in[2];
    const float* Wh = (const float*)d_in[3];
    const float* Wp = (const float*)d_in[4];
    const float* Wx = (const float*)d_in[5];
    const float* w  = (const float*)d_in[6];
    float* out = (float*)d_out;
    char*  ws  = (char*)d_ws;

    // workspace carve-up (needs ~74 MB)
    _Float16* wy16 = (_Float16*)(ws + 0x0000000);
    _Float16* wh16 = (_Float16*)(ws + 0x0040000);
    _Float16* wp16 = (_Float16*)(ws + 0x0080000);
    _Float16* wx16 = (_Float16*)(ws + 0x00C0000);
    float*    wpad = (float*)   (ws + 0x0100000);
    _Float16* hn16 = (_Float16*)(ws + 0x0140000);   // 10.0 MB
    float*    Z1   = (float*)   (ws + 0x1000000);   // 20.0 MB
    float*    Z2   = (float*)   (ws + 0x2800000);   // 20.0 MB
    _Float16* r16  = (_Float16*)(ws + 0x4000000);   // 10.0 MB

    // allow >64KB dynamic LDS for the attention kernel (gfx1250: 320KB/WG)
    (void)hipFuncSetAttribute((const void*)attn_kernel,
                              hipFuncAttributeMaxDynamicSharedMemorySize, DYN_LDS);

    // 1) f32 -> padded f16 conversions
    {
        const long total = (long)BATCH * DP;           // dominant job
        const int blocks = (int)((total + 255) / 256); // 20480
        prep_kernel<<<blocks, 256, 0, stream>>>(Wy, Wh, Wp, Wx, hn, w,
                                                wy16, wh16, wp16, wx16, hn16, wpad);
    }

    // 2) Z1 = W_h h_n^T,  Z2 = W_x h_n^T
    {
        const int waves  = (DP / 16) * (BATCH / 16);   // 20480
        const int blocks = waves * 32 / 256;           // 2560
        gemm_hn_kernel<<<blocks, 256, 0, stream>>>(wh16, hn16, Z1);
        gemm_hn_kernel<<<blocks, 256, 0, stream>>>(wx16, hn16, Z2);
    }

    // 3) fused attention: M, alpha, r
    attn_kernel<<<BATCH / BT, ATTN_THREADS, DYN_LDS, stream>>>(Y, Z1, wy16, wpad, r16);

    // 4) h = tanh(W_p r + Z2)
    {
        const int waves  = (DP / 16) * (BATCH / 16);
        const int blocks = waves * 32 / 256;
        gemm_out_kernel<<<blocks, 256, 0, stream>>>(wp16, r16, Z2, out);
    }
}
